// MultiHeadSelfAttention_74526272520985
// MI455X (gfx1250) — matrix operand order repaired and checked by reading, not compiled, not run on hardware
//
#include <hip/hip_runtime.h>

// MI455X / gfx1250, wave32. B=4, N=4096, D=1024, H=16, HD=64.
// Pipeline: cvt(fp32->bf16) -> QKV GEMM (bf16 WMMA, async-LDS-staged weights)
// -> per-token head-mix attention (bf16 WMMA + fp32 softmax)
// -> proj GEMM (bf16 WMMA, fp32 out + bias).

typedef __attribute__((ext_vector_type(16))) __bf16 v16bf;
typedef __attribute__((ext_vector_type(8)))  __bf16 v8bf;
typedef __attribute__((ext_vector_type(8)))  float  v8f;
typedef int v4i __attribute__((ext_vector_type(4)));

#if __has_builtin(__builtin_amdgcn_global_load_async_to_lds_b128)
#define HAVE_ASYNC_LDS 1
#else
#define HAVE_ASYNC_LDS 0
#endif

__device__ __forceinline__ void wait_async() {
#if HAVE_ASYNC_LDS
#if __has_builtin(__builtin_amdgcn_s_wait_asynccnt)
  __builtin_amdgcn_s_wait_asynccnt(0);
#else
  asm volatile("s_wait_asynccnt 0" ::: "memory");
#endif
#endif
}

__device__ __forceinline__ unsigned short f2bu(float f) {
  unsigned u = __float_as_uint(f);
  unsigned r = u + 0x7FFFu + ((u >> 16) & 1u);   // round-to-nearest-even
  return (unsigned short)(r >> 16);
}
__device__ __forceinline__ __bf16 us2bf(unsigned short s) {
  union { unsigned short u; __bf16 b; } c; c.u = s; return c.b;
}
__device__ __forceinline__ v8f vzero8() {
  v8f z;
#pragma unroll
  for (int i = 0; i < 8; ++i) z[i] = 0.f;
  return z;
}
__device__ __forceinline__ v8f wmma_bf16(v16bf a, v16bf b, v8f c) {
  return __builtin_amdgcn_wmma_f32_16x16x32_bf16(
      false, a, false, b, (short)0, c, false, false);
}

// Copy 64B global -> LDS (4 x b128), async via gfx1250 GLOBAL_LOAD_ASYNC_TO_LDS.
__device__ __forceinline__ void stage_tile(const unsigned short* gB,
                                           unsigned short* sB) {
#pragma unroll
  for (int c = 0; c < 4; ++c) {
#if HAVE_ASYNC_LDS
    __builtin_amdgcn_global_load_async_to_lds_b128(
        (__attribute__((address_space(1))) v4i*)(gB + c * 8),
        (__attribute__((address_space(3))) v4i*)(sB + c * 8), 0, 0);
#else
    *reinterpret_cast<uint4*>(sB + c * 8) =
        *reinterpret_cast<const uint4*>(gB + c * 8);
#endif
  }
}

// ---------------------------------------------------------------- convert
__global__ __launch_bounds__(256)
void cvt_f32_bf16(const float4* __restrict__ src, ushort4* __restrict__ dst, int n4) {
  int i = blockIdx.x * blockDim.x + threadIdx.x;
  if (i < n4) {
    float4 f = src[i];
    ushort4 o;
    o.x = f2bu(f.x); o.y = f2bu(f.y); o.z = f2bu(f.z); o.w = f2bu(f.w);
    dst[i] = o;
  }
}

// ---------------------------------------------------------------- GEMM
// C(MxN) = A(MxK) @ Bw(NxK)^T  [+ bias], row-major bf16 inputs.
// Block: 256 thr = 8 waves; block tile 128x128; wave tile 32x64 (2x4 accs).
// B tile (128 cols x 64 K) double-buffered in LDS via async global->LDS copies.
// Column stride 72 ushorts: 16B-aligned ds_load_b128 + conflict-free banks.
// M,N multiples of 128; K multiple of 64 (true for all call sites).
#define BCOLS 128
#define BSTRIDE 72            // 64 K-elems + 8 pad
#define BUFSZ (BCOLS * BSTRIDE)

template <bool OUT_BF16>
__global__ __launch_bounds__(256)
void gemm_bf16(const unsigned short* __restrict__ A,
               const unsigned short* __restrict__ Bw,
               const float* __restrict__ bias,
               float* __restrict__ Cf,
               unsigned short* __restrict__ Cbf,
               int M, int N, int K) {
  __shared__ unsigned short Bs[2 * BUFSZ];

  const int tid  = threadIdx.x;
  const int lane = tid & 31;
  const int w    = tid >> 5;
  const int lr   = lane & 15;      // row/col within 16-tile
  const int lh   = lane >> 4;      // half-wave select
  const int m0   = blockIdx.y * 128 + (w >> 1) * 32;
  const int n0w  = (w & 1) * 64;   // wave col offset inside block tile
  const int nblk = blockIdx.x * 128;

  // Staging assignment: 2 threads per column, 64B (32 K-elems) each.
  const int scol = tid >> 1;
  const int shf  = (tid & 1) * 32;
  const unsigned short* gbase = Bw + (size_t)(nblk + scol) * K + shf;
  unsigned short* sbase = &Bs[0] + scol * BSTRIDE + shf;

  v8f acc[2][4];
#pragma unroll
  for (int mi = 0; mi < 2; ++mi)
#pragma unroll
    for (int ni = 0; ni < 4; ++ni) acc[mi][ni] = vzero8();

  const unsigned short* aRow[2];
  aRow[0] = A + (size_t)(m0 + lr) * K;
  aRow[1] = A + (size_t)(m0 + 16 + lr) * K;

  stage_tile(gbase, sbase);                       // buffer 0 <- k0=0

  for (int k0 = 0; k0 < K; k0 += 64) {
    const int buf = (k0 >> 6) & 1;
    wait_async();                                  // own async writes done
    __syncthreads();                               // everyone's writes done
    if (k0 + 64 < K)
      stage_tile(gbase + k0 + 64, sbase + (buf ^ 1) * BUFSZ);

    if (k0 + 128 < K) {                            // L2 prefetch of future A
      __builtin_prefetch(aRow[0] + k0 + 128, 0, 0);
      __builtin_prefetch(aRow[1] + k0 + 128, 0, 0);
    }

    const unsigned short* bufp = &Bs[0] + buf * BUFSZ;
#pragma unroll
    for (int ks = 0; ks < 64; ks += 32) {
      v16bf afr[2];
#pragma unroll
      for (int mi = 0; mi < 2; ++mi) {
        // A 16x32 bf16, contiguous k order used here (matches B): lanes0-15 K=0..15, lanes16-31 K=16..31.
        v8bf lo = *reinterpret_cast<const v8bf*>(aRow[mi] + k0 + ks + lh * 16);
        v8bf hi = *reinterpret_cast<const v8bf*>(aRow[mi] + k0 + ks + 8 + lh * 16);
        v16bf a;
#pragma unroll
        for (int i = 0; i < 8; ++i) { a[i] = lo[i]; a[i + 8] = hi[i]; }
        afr[mi] = a;
      }
      v16bf bfr[4];
#pragma unroll
      for (int ni = 0; ni < 4; ++ni) {
        // B 32x16, same contiguous k order as A: lane = column, lanes0-15 K=0..15, lanes16-31 K=16..31.
        bfr[ni] = *reinterpret_cast<const v16bf*>(
            bufp + (n0w + ni * 16 + lr) * BSTRIDE + ks + lh * 16);
      }
#pragma unroll
      for (int mi = 0; mi < 2; ++mi)
#pragma unroll
        for (int ni = 0; ni < 4; ++ni)
          acc[mi][ni] = wmma_bf16(afr[mi], bfr[ni], acc[mi][ni]);
    }
  }

  // Epilogue. C layout: reg r, lane l -> row = r + 8*(l>>4), col = l&15.
#pragma unroll
  for (int mi = 0; mi < 2; ++mi)
#pragma unroll
    for (int ni = 0; ni < 4; ++ni) {
      const int col = nblk + n0w + ni * 16 + lr;
      const float bb = OUT_BF16 ? 0.f : bias[col];
#pragma unroll
      for (int r = 0; r < 8; ++r) {
        const int row = m0 + mi * 16 + r + 8 * lh;
        if (OUT_BF16) {
          Cbf[(size_t)row * N + col] = f2bu(acc[mi][ni][r]);
        } else {
          Cf[(size_t)row * N + col] = acc[mi][ni][r] + bb;
        }
      }
    }
}

// ---------------------------------------------------------------- attention
// One wave per token. qkv row (bf16): [q(16x64) | k(16x64) | v(16x64)].
// S^T = K @ Q^T via WMMA; the C-layout of S^T puts P[h][g] exactly in the
// A-fragment layout needed for O = P @ V (K padded 16->32 with zeros), so
// softmax costs 8 in-register ops + one shfl_xor(16). Output scattered per
// the reference's swapaxes+reshape: row i = h*256 + n/16, col j = (n%16)*64+d.
__global__ __launch_bounds__(256)
void attn_headmix(const unsigned short* __restrict__ qkv,
                  unsigned short* __restrict__ obf) {
  const int lane = threadIdx.x & 31;
  const int w    = threadIdx.x >> 5;
  const int t    = blockIdx.x * 8 + w;       // token = b*4096 + n
  const int b    = t >> 12;
  const int n    = t & 4095;
  const int lr   = lane & 15;
  const int lh   = lane >> 4;

  const unsigned short* qp = qkv + (size_t)t * 3072;
  const unsigned short* kp = qp + 1024;
  const unsigned short* vp = qp + 2048;

  // S^T[g][h] = sum_d K[g][d] * Q[h][d]
  v8f st = vzero8();
#pragma unroll
  for (int kk0 = 0; kk0 < 64; kk0 += 32) {
    v8bf lo = *reinterpret_cast<const v8bf*>(kp + lr * 64 + kk0 + lh * 16);
    v8bf hi = *reinterpret_cast<const v8bf*>(kp + lr * 64 + kk0 + 8 + lh * 16);
    v16bf ak;
#pragma unroll
    for (int i = 0; i < 8; ++i) { ak[i] = lo[i]; ak[i + 8] = hi[i]; }
    v16bf bq = *reinterpret_cast<const v16bf*>(qp + lr * 64 + kk0 + lh * 16);
    st = wmma_bf16(ak, bq, st);
  }

  // Lane l holds S[h=l&15][g = r + 8*(l>>4)] for r=0..7.
  const float scale = 0.125f;                 // 64^-0.5
  float mloc = st[0] * scale;
#pragma unroll
  for (int r = 1; r < 8; ++r) mloc = fmaxf(mloc, st[r] * scale);
  const float mrow = fmaxf(mloc, __shfl_xor(mloc, 16, 32));
  float e[8], s = 0.f;
#pragma unroll
  for (int r = 0; r < 8; ++r) { e[r] = __expf(st[r] * scale - mrow); s += e[r]; }
  s += __shfl_xor(s, 16, 32);
  const float inv = 1.0f / s;

  // P A-fragment: lanes0-15 hold g=0..7 (K0..7), lanes16-31 g=8..15 (K8..15);
  // elems 8..15 (K16..31) zero-padded.
  const __bf16 bz = us2bf(0);
  v16bf pa;
#pragma unroll
  for (int j = 0; j < 8; ++j) pa[j] = us2bf(f2bu(e[j] * inv));
#pragma unroll
  for (int j = 8; j < 16; ++j) pa[j] = bz;

  // O[h][d] = sum_g P[h][g] V[g][d], d in 4 chunks of 16.
#pragma unroll
  for (int ni = 0; ni < 4; ++ni) {
    const int dcol = ni * 16 + lr;
    v16bf vb;
#pragma unroll
    for (int g = 0; g < 16; ++g)
      vb[g] = (g < 8) ? us2bf(vp[(g + 8 * lh) * 64 + dcol]) : bz;  // elems 8..15 (K16..31) zero
    v8f oc = vzero8();
    oc = wmma_bf16(pa, vb, oc);
#pragma unroll
    for (int r = 0; r < 8; ++r) {
      const int h   = r + 8 * lh;
      const int row = (b << 12) + (h << 8) + (n >> 4);
      const int col = ((n & 15) << 6) + dcol;
      obf[(size_t)row * 1024 + col] = f2bu(oc[r]);
    }
  }
}

// ---------------------------------------------------------------- launch
extern "C" void kernel_launch(void* const* d_in, const int* in_sizes, int n_in,
                              void* d_out, int out_size, void* d_ws, size_t ws_size,
                              hipStream_t stream) {
  const float* x     = (const float*)d_in[0];   // 4*4096*1024
  const float* wqkv  = (const float*)d_in[1];   // 3072*1024
  const float* wproj = (const float*)d_in[2];   // 1024*1024
  const float* bproj = (const float*)d_in[3];   // 1024

  const int NX  = 4 * 4096 * 1024;  // 16,777,216
  const int NWQ = 3072 * 1024;      //  3,145,728
  const int NWP = 1024 * 1024;      //  1,048,576

  unsigned short* ws    = (unsigned short*)d_ws;
  unsigned short* xbf   = ws;                          // 16,777,216
  unsigned short* wqkvb = xbf + NX;                    //  3,145,728
  unsigned short* wprojb= wqkvb + NWQ;                 //  1,048,576
  unsigned short* qkvb  = wprojb + NWP;                // 50,331,648
  unsigned short* obf   = qkvb + (size_t)16384 * 3072; // 16,777,216

  // fp32 -> bf16 pre-passes (memory-bound, vectorized 4-wide)
  cvt_f32_bf16<<<(NX / 4 + 255) / 256, 256, 0, stream>>>(
      (const float4*)x, (ushort4*)xbf, NX / 4);
  cvt_f32_bf16<<<(NWQ / 4 + 255) / 256, 256, 0, stream>>>(
      (const float4*)wqkv, (ushort4*)wqkvb, NWQ / 4);
  cvt_f32_bf16<<<(NWP / 4 + 255) / 256, 256, 0, stream>>>(
      (const float4*)wproj, (ushort4*)wprojb, NWP / 4);

  // qkv = x @ w_qkv^T : M=16384, N=3072, K=1024, bf16 out
  gemm_bf16<true><<<dim3(3072 / 128, 16384 / 128), 256, 0, stream>>>(
      xbf, wqkvb, nullptr, nullptr, qkvb, 16384, 3072, 1024);

  // per-token head-mix attention, 1 wave/token, 8 waves/block
  attn_headmix<<<16384 / 8, 256, 0, stream>>>(qkvb, obf);

  // out = o @ w_proj^T + b : M=16384, N=1024, K=1024, fp32 out
  gemm_bf16<false><<<dim3(1024 / 128, 16384 / 128), 256, 0, stream>>>(
      obf, wprojb, bproj, (float*)d_out, nullptr, 16384, 1024, 1024);
}
